// GCN_67731634258670
// MI455X (gfx1250) — matrix operand order repaired and checked by reading, not compiled, not run on hardware
//
#include <hip/hip_runtime.h>
#include <math.h>

// ---------------------------------------------------------------------------
// GNN forward (SAGE-mean conv + BN/ReLU MLP stack + log_softmax) for MI455X.
// N=100000 nodes, E=1.6M edges, 128 in-feats, 256 hidden.
// bf16 WMMA (v_wmma_f32_16x16x32_bf16) for all GEMMs, fp32 accumulate.
// A row-strips staged into LDS via the Tensor Data Mover (tensor_load_to_lds,
// TENSORcnt), read back as WMMA fragments with ds_load_b128.
// BN column stats fused into GEMM epilogue via f32 atomics (L2-resident).
// ---------------------------------------------------------------------------

#define IN_C 128
#define HIDP 256

typedef __attribute__((ext_vector_type(16))) __bf16          v16bf;
typedef __attribute__((ext_vector_type(8)))  float           v8f;
typedef __attribute__((ext_vector_type(8)))  unsigned short  u16x8;
typedef __attribute__((ext_vector_type(4)))  unsigned int    v4u;
typedef __attribute__((ext_vector_type(8)))  int             v8i;
typedef __attribute__((ext_vector_type(4)))  int             v4i;

static __device__ __forceinline__ unsigned short f2bf(float f) {
  unsigned u = __float_as_uint(f);
  u += 0x7FFFu + ((u >> 16) & 1u);          // round-to-nearest-even
  return (unsigned short)(u >> 16);
}
static __device__ __forceinline__ float bf2f(unsigned short h) {
  return __uint_as_float(((unsigned)h) << 16);
}

// ---------------------------- zero fill ------------------------------------
__global__ void zero_f4(float4* __restrict__ p, size_t n4) {
  size_t i = (size_t)blockIdx.x * blockDim.x + threadIdx.x;
  if (i < n4) p[i] = make_float4(0.f, 0.f, 0.f, 0.f);
}

// ------------------------ edge scatter (mean agg) --------------------------
// One wave per edge: 32 lanes x 4 floats = 128 feature atomics. x and agg
// both fit in the 192MB L2, so these atomics stay on-die.
__global__ __launch_bounds__(256) void scatter_edges(
    const long long* __restrict__ ei, const float* __restrict__ x,
    float* __restrict__ agg, float* __restrict__ cnt, int E) {
  int lane = threadIdx.x & 31;
  int e = blockIdx.x * 8 + (threadIdx.x >> 5);
  if (e >= E) return;
  long long s = ei[e];
  long long d = ei[(size_t)E + e];
  const float* xs = x + (size_t)s * IN_C;
  float*       ad = agg + (size_t)d * IN_C;
#pragma unroll
  for (int j = 0; j < IN_C / 32; ++j)
    atomicAdd(&ad[lane + 32 * j], xs[lane + 32 * j]);
  if (lane == 0) atomicAdd(&cnt[d], 1.0f);
}

// --------------- build A0 = [agg/max(cnt,1) | x] as bf16 [N,256] -----------
__global__ void build_A0(const float* __restrict__ x, const float* __restrict__ agg,
                         const float* __restrict__ cnt, unsigned short* __restrict__ A0,
                         int Nrows) {
  size_t i = (size_t)blockIdx.x * blockDim.x + threadIdx.x;
  size_t node = i >> 8;
  int c = (int)(i & 255);
  if (node >= (size_t)Nrows) return;
  float v;
  if (c < IN_C) {
    float inv = 1.0f / fmaxf(cnt[node], 1.0f);
    v = agg[node * IN_C + c] * inv;
  } else {
    v = x[node * IN_C + (c - IN_C)];
  }
  A0[i] = f2bf(v);
}

// --------------- weight prep: Wcat = [Wl | Wr] -> bf16 [256,256] -----------
__global__ void build_Wcat(const float* __restrict__ Wl, const float* __restrict__ Wr,
                           unsigned short* __restrict__ Wc) {
  int i = blockIdx.x * blockDim.x + threadIdx.x;   // 256*256 threads
  int o = i >> 8, k = i & 255;
  float v = (k < IN_C) ? Wl[o * IN_C + k] : Wr[o * IN_C + (k - IN_C)];
  Wc[i] = f2bf(v);
}

// --------------- generic weight convert with zero row padding --------------
__global__ void cvt_w(const float* __restrict__ W, unsigned short* __restrict__ Wb,
                      int Rreal, int Rp, int K) {
  int i = blockIdx.x * blockDim.x + threadIdx.x;
  if (i >= Rp * K) return;
  int r = i / K, k = i - r * K;
  Wb[i] = f2bf((r < Rreal) ? W[r * K + k] : 0.0f);
}

// --------------------- GEMM epilogue: bias + BN stats ----------------------
static __device__ __forceinline__ void gemm_epilogue(
    v8f acc, int rowBase, int colTile, int m, int half,
    const float* __restrict__ bias, int biasLimit,
    float* __restrict__ H, float* __restrict__ stats, int Coutp) {
  int col = colTile * 16 + m;
  float bv = (col < biasLimit) ? bias[col] : 0.0f;
  float s = 0.f, q = 0.f;
  float* Hb = H + (size_t)rowBase * Coutp + col;
#pragma unroll
  for (int r = 0; r < 8; ++r) {                  // rows M = r + half*8
    float v = acc[r] + bv;
    Hb[(size_t)(r + half * 8) * Coutp] = v;
    s += v;
    q += v * v;
  }
  atomicAdd(&stats[col], s);
  atomicAdd(&stats[Coutp + col], q);
}

// ---------------------------------------------------------------------------
// Fused GEMM + bias + BN-stat epilogue with TDM-staged A operand.
//   H[n,o] = sum_k A[n,k]*B[o,k] + bias[o]; A bf16 [N,K], B bf16 [Coutp,K].
// Block = 8 waves, owns a (RTB*16)-row strip of A, DMA'd once into LDS by the
// Tensor Data Mover (rows beyond N auto-zero-filled: tensor_dim1 = N).
// Wave map (compile-time RTB x CPW): localRowTile = wave / (8/RTB); each wave
// computes CPW adjacent 16x16 col tiles sharing one A fragment per K-step
// (2x ds_load_b128). B fragment = two 16-byte chunks of the weight row
// (o = colBase + lane&15) at K = kb + half*8 and kb + 16 + half*8 (ISA 16x16x32 bf16 B layout).
// ---------------------------------------------------------------------------
template <int RTB, int CPW>
__global__ __launch_bounds__(256) void gemm_bn(
    const unsigned short* __restrict__ A, const unsigned short* __restrict__ B,
    const float* __restrict__ bias, int biasLimit,
    float* __restrict__ H, float* __restrict__ stats,
    int Nrows, int K, int Coutp) {
  extern __shared__ unsigned short smem[];            // (RTB*16) x K bf16
  constexpr int wavesPerRow  = 8 / RTB;               // colTiles == wavesPerRow*CPW
  constexpr int rowsPerBlock = RTB * 16;
  const int lane   = threadIdx.x & 31;
  const int waveId = threadIdx.x >> 5;
  const int rowBase0 = blockIdx.x * rowsPerBlock;

#if __has_builtin(__builtin_amdgcn_tensor_load_to_lds)
  if (threadIdx.x < 32) {                        // wave 0 issues the TDM DMA
    unsigned lds_base = (unsigned)(size_t)&smem[0];      // LDS byte offset
    size_t ga = (size_t)A + (size_t)rowBase0 * K * 2;    // tile start address
    v4u g0 = { 1u,                                       // count=1, user mode
               lds_base,
               (unsigned)(ga & 0xffffffffull),
               (unsigned)((ga >> 32) & 0x1ffffffull) | 0x80000000u };  // type=2
    unsigned t0 = (unsigned)K, t1 = (unsigned)Nrows;
    v8i g1 = { (int)(1u << 16),                                  // data_size=2B
               (int)((t0 & 0xffffu) << 16),                      // tdim0[15:0]
               (int)(((t0 >> 16) & 0xffffu) | ((t1 & 0xffffu) << 16)),
               (int)(((t1 >> 16) & 0xffffu) | ((t0 & 0xffffu) << 16)),  // tile_dim0=K
               (int)(unsigned)rowsPerBlock,                      // tile_dim1
               (int)t0,                                          // dim0_stride=K
               0, 0 };
    v4i gz = { 0, 0, 0, 0 };
#if defined(__clang_major__) && (__clang_major__ >= 23)
    v8i gz8 = { 0, 0, 0, 0, 0, 0, 0, 0 };
    __builtin_amdgcn_tensor_load_to_lds(g0, g1, gz, gz, gz8, 0);
#else
    __builtin_amdgcn_tensor_load_to_lds(g0, g1, gz, gz, 0);
#endif
    __builtin_amdgcn_s_wait_tensorcnt(0);
  }
  __syncthreads();
#else
  // Cooperative fallback: 16B chunks, zero-pad rows beyond N.
  for (int idx = threadIdx.x; idx < rowsPerBlock * (K / 8); idx += 256) {
    int r = idx / (K / 8), c8 = idx - r * (K / 8);
    int gr = rowBase0 + r;
    u16x8 v = {};
    if (gr < Nrows) v = *(const u16x8*)(A + (size_t)gr * K + c8 * 8);
    *(u16x8*)(smem + (size_t)r * K + c8 * 8) = v;
  }
  __syncthreads();
#endif

  const int localRowTile = waveId / wavesPerRow;
  const int colTileBase  = (waveId - localRowTile * wavesPerRow) * CPW;
  const int m = lane & 15;
  const int half = lane >> 4;
  const int rowBase = rowBase0 + localRowTile * 16;
  if (rowBase >= Nrows) return;                  // wave-uniform tail guard

  const unsigned short* lrow = smem + (size_t)(localRowTile * 16 + m) * K;
  const unsigned short* Brow[CPW];
#pragma unroll
  for (int c = 0; c < CPW; ++c)
    Brow[c] = B + (size_t)((colTileBase + c) * 16 + m) * K;

  v8f acc[CPW];
#pragma unroll
  for (int c = 0; c < CPW; ++c) acc[c] = (v8f){};

  for (int kb = 0; kb < K; kb += 32) {
    union { u16x8 h[2]; v16bf v; } a, b[CPW];
    a.h[0] = *(const u16x8*)(lrow + kb + half * 8);        // ds_load_b128
    a.h[1] = *(const u16x8*)(lrow + kb + 16 + half * 8);
#pragma unroll
    for (int c = 0; c < CPW; ++c) {
      b[c].h[0] = *(const u16x8*)(Brow[c] + kb + half * 8);
      b[c].h[1] = *(const u16x8*)(Brow[c] + kb + 16 + half * 8);
    }
    if (kb + 32 < K)
      __builtin_prefetch((const void*)(Brow[0] + kb + 32), 0, 3);
#pragma unroll
    for (int c = 0; c < CPW; ++c)
      acc[c] = __builtin_amdgcn_wmma_f32_16x16x32_bf16(
          false, a.v, false, b[c].v, (short)0, acc[c], false, false);
  }
#pragma unroll
  for (int c = 0; c < CPW; ++c)
    gemm_epilogue(acc[c], rowBase, colTileBase + c, m, half, bias, biasLimit, H, stats, Coutp);
}

// ------------- per-column BN params: scale = g*rsqrt(var+eps), shift -------
__global__ void bn_params(const float* __restrict__ stats, const float* __restrict__ gamma,
                          const float* __restrict__ beta, int Creal, int Coutp,
                          float invN, float* __restrict__ ss) {
  int c = threadIdx.x;
  if (c >= Coutp) return;
  if (c >= Creal) { ss[c] = 0.f; ss[Coutp + c] = 0.f; return; }  // padded cols -> 0
  float m  = stats[c] * invN;
  float v  = stats[Coutp + c] * invN - m * m;
  float sc = gamma[c] * rsqrtf(v + 1e-5f);
  ss[c] = sc;
  ss[Coutp + c] = beta[c] - m * sc;
}

// ------------- apply BN + ReLU, convert to bf16 for next GEMM --------------
__global__ void apply_bn_relu(const float4* __restrict__ H, const float* __restrict__ ss,
                              ushort4* __restrict__ Aout, int mask, int Coutp, size_t nvec) {
  size_t i = (size_t)blockIdx.x * blockDim.x + threadIdx.x;
  if (i >= nvec) return;
  float4 h = H[i];
  int c0 = (int)((i * 4) & (size_t)mask);             // Coutp is a power of two
  ushort4 o;
  o.x = f2bf(fmaxf(h.x * ss[c0 + 0] + ss[Coutp + c0 + 0], 0.f));
  o.y = f2bf(fmaxf(h.y * ss[c0 + 1] + ss[Coutp + c0 + 1], 0.f));
  o.z = f2bf(fmaxf(h.z * ss[c0 + 2] + ss[Coutp + c0 + 2], 0.f));
  o.w = f2bf(fmaxf(h.w * ss[c0 + 3] + ss[Coutp + c0 + 3], 0.f));
  Aout[i] = o;
}

// ------------- head: [N,14(pad16)] @ W4^T + b4, then log_softmax -----------
__global__ void head_logsoftmax(const unsigned short* __restrict__ A3,
                                const float* __restrict__ W4, const float* __restrict__ b4,
                                float* __restrict__ out, int Nrows) {
  int i = blockIdx.x * blockDim.x + threadIdx.x;
  if (i >= Nrows) return;
  const unsigned short* row = A3 + (size_t)i * 16;
  float z0 = b4[0], z1 = b4[1];
#pragma unroll
  for (int k = 0; k < 14; ++k) {
    float a = bf2f(row[k]);
    z0 += a * W4[k];
    z1 += a * W4[14 + k];
  }
  float mx = fmaxf(z0, z1);
  float l  = logf(__expf(z0 - mx) + __expf(z1 - mx));
  out[i * 2 + 0] = z0 - mx - l;
  out[i * 2 + 1] = z1 - mx - l;
}

// ---------------------------------------------------------------------------
extern "C" void kernel_launch(void* const* d_in, const int* in_sizes, int n_in,
                              void* d_out, int out_size, void* d_ws, size_t ws_size,
                              hipStream_t stream) {
  (void)n_in; (void)out_size; (void)ws_size;
  const float*     x   = (const float*)d_in[0];
  const long long* ei  = (const long long*)d_in[1];
  const float* Wl = (const float*)d_in[2];
  const float* bl = (const float*)d_in[3];
  const float* Wr = (const float*)d_in[4];
  const float* bng = (const float*)d_in[5];
  const float* bnb = (const float*)d_in[6];
  const float* W1 = (const float*)d_in[7];
  const float* b1 = (const float*)d_in[8];
  const float* W2 = (const float*)d_in[9];
  const float* b2 = (const float*)d_in[10];
  const float* W3 = (const float*)d_in[11];
  const float* b3 = (const float*)d_in[12];
  const float* W4 = (const float*)d_in[13];
  const float* b4 = (const float*)d_in[14];
  const float* g1 = (const float*)d_in[15];  const float* bb1 = (const float*)d_in[16];
  const float* g2 = (const float*)d_in[17];  const float* bb2 = (const float*)d_in[18];
  const float* g3 = (const float*)d_in[19];  const float* bb3 = (const float*)d_in[20];

  const int N = in_sizes[0] / IN_C;   // 100000 (multiple of 16)
  const int E = in_sizes[1] / 2;      // 1600000
  const float invN = 1.0f / (float)N;

  // --- workspace carve-out (aligned 256B). agg overlaps the start of H:
  // agg is fully consumed by build_A0 before GEMM1 writes H over it.
  char* ws = (char*)d_ws;
  size_t cur = 0;
  auto carve = [&](size_t bytes) -> char* {
    char* p = ws + cur;
    cur += (bytes + 255) & ~(size_t)255;
    return p;
  };
  float*          H     = (float*)carve((size_t)N * HIDP * sizeof(float));
  float*          agg   = H;  // first N*128 floats of H region
  unsigned short* Aping = (unsigned short*)carve((size_t)N * HIDP * 2);
  unsigned short* Apong = (unsigned short*)carve((size_t)N * HIDP * 2);
  float*          cnt   = (float*)carve((size_t)N * sizeof(float));
  unsigned short* Wc    = (unsigned short*)carve(256 * 256 * 2);
  unsigned short* W1b   = (unsigned short*)carve(256 * 256 * 2);
  unsigned short* W2b   = (unsigned short*)carve(64 * 256 * 2);
  unsigned short* W3b   = (unsigned short*)carve(16 * 64 * 2);
  float*          stats = (float*)carve(4 * 512 * sizeof(float));
  float*          ss    = (float*)carve(4 * 512 * sizeof(float));

  auto zero = [&](float* p, size_t n) {   // n divisible by 4
    size_t n4 = n / 4;
    zero_f4<<<(unsigned)((n4 + 255) / 256), 256, 0, stream>>>((float4*)p, n4);
  };
  zero(agg,   (size_t)N * IN_C);
  zero(cnt,   (size_t)N);
  zero(stats, 4 * 512);

  // --- SAGE mean aggregation
  scatter_edges<<<(E + 7) / 8, 256, 0, stream>>>(ei, x, agg, cnt, E);
  build_A0<<<(unsigned)(((size_t)N * 256 + 255) / 256), 256, 0, stream>>>(x, agg, cnt, Aping, N);

  // --- weight conversion (tiny, L2-resident afterwards)
  build_Wcat<<<256, 256, 0, stream>>>(Wl, Wr, Wc);
  cvt_w<<<256, 256, 0, stream>>>(W1, W1b, 256, 256, 256);
  cvt_w<<<64, 256, 0, stream>>>(W2, W2b, 64, 64, 256);
  cvt_w<<<4, 256, 0, stream>>>(W3, W3b, 14, 16, 64);

  auto apply = [&](const float* stA, const float* gamma, const float* beta, int Creal,
                   int Coutp, float* ssL, unsigned short* Aout) {
    bn_params<<<1, 256, 0, stream>>>(stA, gamma, beta, Creal, Coutp, invN, ssL);
    size_t nvec = (size_t)N * Coutp / 4;
    apply_bn_relu<<<(unsigned)((nvec + 255) / 256), 256, 0, stream>>>(
        (const float4*)H, ssL, (ushort4*)Aout, Coutp - 1, Coutp, nvec);
  };

  // L0: [agg|x] @ Wcat^T + bl -> BN(bn_g,bn_b) -> ReLU     (256 -> 256)
  gemm_bn<1, 2><<<(N + 15) / 16, 256, 16 * 256 * 2, stream>>>(
      Aping, Wc, bl, 256, H, stats + 0 * 512, N, 256, 256);
  apply(stats + 0 * 512, bng, bnb, 256, 256, ss + 0 * 512, Apong);
  // L1: @ W1^T + b1 -> BN(g1,bb1) -> ReLU                  (256 -> 256)
  gemm_bn<1, 2><<<(N + 15) / 16, 256, 16 * 256 * 2, stream>>>(
      Apong, W1b, b1, 256, H, stats + 1 * 512, N, 256, 256);
  apply(stats + 1 * 512, g1, bb1, 256, 256, ss + 1 * 512, Aping);
  // L2: @ W2^T + b2 -> BN(g2,bb2) -> ReLU                  (256 -> 64)
  gemm_bn<2, 1><<<(N + 31) / 32, 256, 32 * 256 * 2, stream>>>(
      Aping, W2b, b2, 64, H, stats + 2 * 512, N, 256, 64);
  apply(stats + 2 * 512, g2, bb2, 64, 64, ss + 2 * 512, Apong);
  // L3: @ W3^T + b3 -> BN(g3,bb3) -> ReLU                  (64 -> 14 pad 16)
  gemm_bn<8, 1><<<(N + 127) / 128, 256, 128 * 64 * 2, stream>>>(
      Apong, W3b, b3, 14, H, stats + 3 * 512, N, 64, 16);
  apply(stats + 3 * 512, g3, bb3, 14, 16, ss + 3 * 512, Aping);
  // head: 14 -> 2 + log_softmax
  head_logsoftmax<<<(N + 255) / 256, 256, 0, stream>>>(Aping, W4, b4, (float*)d_out, N);
}
